// WaveletLayer_73186242724063
// MI455X (gfx1250) — matrix operand order repaired and checked by reading, not compiled, not run on hardware
//
#include <hip/hip_runtime.h>
#include <math.h>
#include <stdint.h>

typedef __attribute__((ext_vector_type(16))) _Float16 v16h;
typedef __attribute__((ext_vector_type(8)))  _Float16 v8h;
typedef __attribute__((ext_vector_type(4)))  _Float16 v4h;
typedef __attribute__((ext_vector_type(8)))  float    v8f;

// db2 filter taps (h0..h3); DEC_LO reversed == h, DEC_HI reversed == REC_HI
#define H0C 0.48296291314469025f
#define H1C 0.83651630373780790f
#define H2C 0.22414386804185735f
#define H3C (-0.12940952255092145f)

#define BLK_M 128
#define BLK_N 128
#define BLK_K 32
#define LDSK  40   // padded row stride in halves (80B, multiple of 16B)

// CDNA5 async global->LDS path (guarded; falls back to sync fill if absent)
#if defined(__AMDGCN__) && __has_builtin(__builtin_amdgcn_global_load_async_to_lds_b128) && __has_builtin(__builtin_amdgcn_s_wait_asynccnt)
#define USE_ASYNC_LDS 1
typedef int v4i_vec __attribute__((vector_size(16)));
typedef __attribute__((address_space(1))) v4i_vec as1_v4i;
typedef __attribute__((address_space(3))) v4i_vec as3_v4i;
#else
#define USE_ASYNC_LDS 0
#endif

// ---------------------------------------------------------------------------
// Weight prep: w[H,H,3] (o,i,tap) -> wt[o][tap*H + i] as f16, for both convs
// ---------------------------------------------------------------------------
__global__ void __launch_bounds__(256)
prep_conv_w_k(const float* __restrict__ wa, const float* __restrict__ wd,
              _Float16* __restrict__ wta, _Float16* __restrict__ wtd)
{
    const int idx = blockIdx.x * 256 + threadIdx.x;     // [0, 1024*3072)
    const int o   = idx / 3072;
    const int kk  = idx - o * 3072;
    const int tap = kk >> 10;
    const int i   = kk & 1023;
    const int src = o * 3072 + i * 3 + tap;
    wta[idx] = (_Float16)wa[src];
    wtd[idx] = (_Float16)wd[src];
}

__global__ void __launch_bounds__(256)
prep_out_w_k(const float* __restrict__ wo, _Float16* __restrict__ wo16)
{
    const int idx = blockIdx.x * 256 + threadIdx.x;     // [0, 1024*1024)
    wo16[idx] = (_Float16)wo[idx];
}

// ---------------------------------------------------------------------------
// DWT: x[8,2048,1024] f32 -> ca,cd [8,1024,1024] f16 (already truncated to S/2)
// ---------------------------------------------------------------------------
__global__ void __launch_bounds__(256)
dwt_k(const float* __restrict__ x, _Float16* __restrict__ ca, _Float16* __restrict__ cd)
{
    const int bm = blockIdx.x;               // b*1024 + m
    const int b  = bm >> 10, m = bm & 1023;
    const int h  = threadIdx.x << 2;
    const float hl[4] = { H0C,  H1C, H2C,  H3C };
    const float gl[4] = { H3C, -H2C, H1C, -H0C };
    float ax = 0.f, ay = 0.f, az = 0.f, aw = 0.f;
    float dx = 0.f, dy = 0.f, dz = 0.f, dw = 0.f;
#pragma unroll
    for (int t = 0; t < 4; ++t) {
        int i = 2 * m + t - 2;
        if (i < 0) i = -1 - i;               // symmetric left reflection only
        const float4 v = *(const float4*)(x + (((size_t)b * 2048 + (size_t)i) << 10) + h);
        ax += hl[t] * v.x; ay += hl[t] * v.y; az += hl[t] * v.z; aw += hl[t] * v.w;
        dx += gl[t] * v.x; dy += gl[t] * v.y; dz += gl[t] * v.z; dw += gl[t] * v.w;
    }
    v4h va; va[0] = (_Float16)ax; va[1] = (_Float16)ay; va[2] = (_Float16)az; va[3] = (_Float16)aw;
    v4h vd; vd[0] = (_Float16)dx; vd[1] = (_Float16)dy; vd[2] = (_Float16)dz; vd[3] = (_Float16)dw;
    *(v4h*)(ca + ((size_t)bm << 10) + h) = va;
    *(v4h*)(cd + ((size_t)bm << 10) + h) = vd;
}

// ---------------------------------------------------------------------------
// iDWT: pa,pd [8,1024,1024] f16 -> recon [8,2048,1024] f16 (rows 2046/2047 = 0)
// ---------------------------------------------------------------------------
__global__ void __launch_bounds__(256)
idwt_k(const _Float16* __restrict__ pa, const _Float16* __restrict__ pd,
       _Float16* __restrict__ rec)
{
    const int bs = blockIdx.x;               // b*2048 + s
    const int b  = bs >> 11, s = bs & 2047;
    const int h  = threadIdx.x << 2;
    _Float16* dst = rec + ((size_t)bs << 10) + h;
    if (s >= 2046) { v4h z = {}; *(v4h*)dst = z; return; }

    float fa0, fa1, fd0, fd1;
    if ((s & 1) == 0) { fa0 = H2C; fa1 = H0C; fd0 = H1C;  fd1 = H3C;  }
    else              { fa0 = H3C; fa1 = H1C; fd0 = -H0C; fd1 = -H2C; }
    const int i0 = s >> 1;

    const size_t r0 = (((size_t)b << 10) + (size_t)i0) << 10;
    const size_t r1 = r0 + 1024;
    const v4h a0 = *(const v4h*)(pa + r0 + h);
    const v4h a1 = *(const v4h*)(pa + r1 + h);
    const v4h d0 = *(const v4h*)(pd + r0 + h);
    const v4h d1 = *(const v4h*)(pd + r1 + h);
    v4h o;
#pragma unroll
    for (int c = 0; c < 4; ++c) {
        const float y = fa0 * (float)a0[c] + fa1 * (float)a1[c]
                      + fd0 * (float)d0[c] + fd1 * (float)d1[c];
        o[c] = (_Float16)y;
    }
    *(v4h*)dst = o;
}

// ---------------------------------------------------------------------------
// WMMA GEMM.  MODE 0: conv-as-GEMM (A gathered with 3-tap halo from cx
// [B*1024,1024], Ktot=3072) + bias + exact GELU, f16 out.
// MODE 1: plain A [M,1024] (Ktot=1024) + bias, f32 out.
// Block tile 128x128, 8 waves (4Mx2N), wave tile 32x64 -> 8 WMMAs / K-step.
// A tile: sync fill (needs predicated zero halo).  B tile: async->LDS.
// ---------------------------------------------------------------------------
template<int MODE>
__global__ void __launch_bounds__(256)
gemm_wmma_k(const _Float16* __restrict__ A, const _Float16* __restrict__ Bw,
            const float* __restrict__ bias, void* __restrict__ dstv, int Ktot)
{
    __shared__ _Float16 As[BLK_M][LDSK];   // [m][k], rows 16B-aligned
    __shared__ _Float16 Bs[BLK_N][LDSK];   // [n][k] (N-major = transposed B)

    const int tid  = threadIdx.x;
    const int lane = tid & 31;
    const int wave = tid >> 5;
    const int wm   = wave & 3;             // M group 0..3 (32 rows each)
    const int wn   = wave >> 2;            // N group 0..1 (64 cols each)
    const int r    = lane & 15;
    const int hv   = lane >> 4;            // half-wave 0/1

    const int gx = blockIdx.x;             // N tile
    const int gy = blockIdx.y;             // M tile

    // cooperative fill mappings (both A and B: row = tid&127, 32B per thread)
    const int frow  = tid & 127;
    const int fhalf = tid >> 7;            // 0/1 -> which 32B of the 64B row
    const int mg    = gy * BLK_M + frow;   // global M row for A
    const _Float16* bsrc = Bw + (size_t)(gx * BLK_N + frow) * (size_t)Ktot + fhalf * 16;
    _Float16* bdst = &Bs[frow][fhalf * 16];

    v8f acc[2][4] = {};

    const int ksteps = Ktot >> 5;
    for (int kt = 0; kt < ksteps; ++kt) {
        const int kc = kt << 5;
        __syncthreads();
        // ---- A tile: sync fill (with conv halo / zero pad for MODE 0) ----
        {
            uint4 d0 = make_uint4(0u, 0u, 0u, 0u), d1 = d0;
            if (MODE == 0) {
                const int bb  = mg >> 10;
                const int t   = mg & 1023;
                const int tap = kc >> 10;
                const int i0  = kc & 1023;
                const int tt  = t + tap - 1;
                if (tt >= 0 && tt < 1024) {
                    const _Float16* src =
                        A + ((((size_t)bb << 10) + (size_t)tt) << 10) + i0 + fhalf * 16;
                    d0 = *(const uint4*)src;
                    d1 = *(const uint4*)(src + 8);
                }
            } else {
                const _Float16* src = A + ((size_t)mg << 10) + kc + fhalf * 16;
                d0 = *(const uint4*)src;
                d1 = *(const uint4*)(src + 8);
            }
            *(uint4*)&As[frow][fhalf * 16]     = d0;
            *(uint4*)&As[frow][fhalf * 16 + 8] = d1;
        }
        // ---- B tile: async global->LDS (2 x b128 per thread) ----
#if USE_ASYNC_LDS
        __builtin_amdgcn_global_load_async_to_lds_b128(
            (as1_v4i*)(uintptr_t)(bsrc + kc),     (as3_v4i*)bdst,       0, 0);
        __builtin_amdgcn_global_load_async_to_lds_b128(
            (as1_v4i*)(uintptr_t)(bsrc + kc + 8), (as3_v4i*)(bdst + 8), 0, 0);
#else
        {
            const uint4 e0 = *(const uint4*)(bsrc + kc);
            const uint4 e1 = *(const uint4*)(bsrc + kc + 8);
            *(uint4*)bdst       = e0;
            *(uint4*)(bdst + 8) = e1;
        }
#endif
        if (kt + 1 < ksteps) __builtin_prefetch(bsrc + kc + 32, 0, 1);
#if USE_ASYNC_LDS
        __builtin_amdgcn_s_wait_asynccnt(0);
#endif
        __syncthreads();

        // ---- fragments per ISA 16-bit layouts ----
        // A lane: row M=r, K = 8*hv + {0..7} and 8*hv + 16 + {0..7}
        // B lane: col N=r, K = 8*hv + {0..7} and 8*hv + 16 + {0..7}
        v16h af[2], bf[4];
#pragma unroll
        for (int im = 0; im < 2; ++im) {
            const _Float16* p = &As[wm * 32 + im * 16 + r][hv * 8];
            const v8h lo = *(const v8h*)p;
            const v8h hi = *(const v8h*)(p + 16);
            af[im] = __builtin_shufflevector(lo, hi, 0,1,2,3,4,5,6,7,8,9,10,11,12,13,14,15);
        }
#pragma unroll
        for (int in = 0; in < 4; ++in) {
            const _Float16* p = &Bs[wn * 64 + in * 16 + r][hv * 8];
            const v8h lo = *(const v8h*)p;
            const v8h hi = *(const v8h*)(p + 16);
            bf[in] = __builtin_shufflevector(lo, hi, 0,1,2,3,4,5,6,7,8,9,10,11,12,13,14,15);
        }
#pragma unroll
        for (int im = 0; im < 2; ++im)
#pragma unroll
            for (int in = 0; in < 4; ++in)
                acc[im][in] = __builtin_amdgcn_wmma_f32_16x16x32_f16(
                    false, af[im], false, bf[in], (short)0, acc[im][in], false, false);
    }

    // ---- epilogue: element v of acc: row m0+8*hv+v, col n0+r ----
#pragma unroll
    for (int im = 0; im < 2; ++im) {
#pragma unroll
        for (int in = 0; in < 4; ++in) {
            const int col = gx * BLK_N + wn * 64 + in * 16 + r;
            const int m0  = gy * BLK_M + wm * 32 + im * 16 + hv * 8;
            const float bv = bias[col];
#pragma unroll
            for (int v = 0; v < 8; ++v) {
                float y = acc[im][in][v] + bv;
                const size_t off = ((size_t)(m0 + v) << 10) + (size_t)col;
                if (MODE == 0) {
                    y = 0.5f * y * (1.0f + erff(y * 0.70710678118654752440f));
                    ((_Float16*)dstv)[off] = (_Float16)y;
                } else {
                    ((float*)dstv)[off] = y;
                }
            }
        }
    }
}

// ---------------------------------------------------------------------------
extern "C" void kernel_launch(void* const* d_in, const int* in_sizes, int n_in,
                              void* d_out, int out_size, void* d_ws, size_t ws_size,
                              hipStream_t stream)
{
    (void)in_sizes; (void)n_in; (void)out_size; (void)ws_size;
    const float* x  = (const float*)d_in[0];   // [8,2048,1024]
    const float* wa = (const float*)d_in[1];   // [1024,1024,3]
    const float* ba = (const float*)d_in[2];   // [1024]
    const float* wd = (const float*)d_in[3];   // [1024,1024,3]
    const float* bd = (const float*)d_in[4];   // [1024]
    const float* wo = (const float*)d_in[5];   // [1024,1024]
    const float* bo = (const float*)d_in[6];   // [1024]
    float* out = (float*)d_out;                // [8,2048,1024] f32

    const size_t QM = (size_t)8 * 1024 * 1024; // 8M halves per [8,1024,1024] f16 buf
    _Float16* ca16  = (_Float16*)d_ws;
    _Float16* cd16  = ca16  + QM;
    _Float16* pa16  = cd16  + QM;
    _Float16* pd16  = pa16  + QM;
    _Float16* rec16 = pd16  + QM;              // [8,2048,1024] = 16M halves
    _Float16* wta   = rec16 + 2 * QM;
    _Float16* wtd   = wta   + (size_t)1024 * 3072;
    _Float16* wo16  = wtd   + (size_t)1024 * 3072;

    prep_conv_w_k<<<12288, 256, 0, stream>>>(wa, wd, wta, wtd);
    prep_out_w_k <<<4096,  256, 0, stream>>>(wo, wo16);
    dwt_k        <<<8192,  256, 0, stream>>>(x, ca16, cd16);
    // conv GEMMs: M=8192 (64 tiles of 128), N=1024 (8 tiles of 128), K=3072
    gemm_wmma_k<0><<<dim3(8, 64),  256, 0, stream>>>(ca16, wta, ba, (void*)pa16, 3072);
    gemm_wmma_k<0><<<dim3(8, 64),  256, 0, stream>>>(cd16, wtd, bd, (void*)pd16, 3072);
    idwt_k       <<<16384, 256, 0, stream>>>(pa16, pd16, rec16);
    // output GEMM: M=16384 (128 tiles), N=1024 (8 tiles), K=1024
    gemm_wmma_k<1><<<dim3(8, 128), 256, 0, stream>>>(rec16, wo16, bo, (void*)out, 1024);
}
